// GNNEncoder_6528350290280
// MI455X (gfx1250) — matrix operand order repaired and checked by reading, not compiled, not run on hardware
//
#include <hip/hip_runtime.h>
#include <hip/hip_bf16.h>

// ---------------------------------------------------------------------------
// GNN encoder for MI455X (gfx1250, wave32, WMMA bf16 16x16x32, f32 accum).
// Dominant edge-MLP GEMM fused as [E, 8256] @ [8256, 64]: each A-row is
// outer(u[e], h[src[e]]) built in registers (b2 folded as 64 extra K rows).
// B operand is streamed into LDS with global_load_async_to_lds_b128 double
// buffering (43 stages x 6 K-chunks), gated by s_wait_asynccnt.
// ---------------------------------------------------------------------------

constexpr int Nn   = 16384;
constexpr int Ee   = 65536;
constexpr int Dd   = 64;
constexpr int Hh   = 128;
constexpr int Ll   = 3;
constexpr int Gg   = 512;
constexpr int OUTd = 128;

#define LRELU 0.01f
#define LN_EPS 1e-5f

typedef __bf16 bf16_t;
typedef bf16_t v16bf __attribute__((ext_vector_type(16)));
typedef float  v8f   __attribute__((ext_vector_type(8)));

__device__ __forceinline__ v8f wmma_bf16(v16bf a, v16bf b, v8f c) {
    return __builtin_amdgcn_wmma_f32_16x16x32_bf16(false, a, false, b, (short)0, c,
                                                   false, false);
}

// Raw 32-bit LDS byte offset of a generic pointer that points into LDS.
__device__ __forceinline__ unsigned lds_off_of(const void* p) {
    return (unsigned)(unsigned long long)(const __attribute__((address_space(3))) char*)p;
}

// Issue one per-wave async 16B/lane copy: LDS[lds_addr] = MEM[gaddr].
__device__ __forceinline__ void async_copy_b128(unsigned lds_addr, const void* gaddr) {
    asm volatile("global_load_async_to_lds_b128 %0, %1, off"
                 :: "v"(lds_addr), "v"(gaddr)
                 : "memory");
}

// A fragment (16x32 bf16) from a row-major LDS row pointer.
// Lane l: m = l&15, hi = l>>4.  elems j<8: K = kbase + hi*8 + j
//                               elems j>=8: K = kbase + 16 + hi*8 + (j-8)
__device__ __forceinline__ v16bf a_frag_row(const bf16_t* row, int kbase, int hi) {
    v16bf a;
#pragma unroll
    for (int j = 0; j < 8; ++j) a[j] = row[kbase + hi * 8 + j];
#pragma unroll
    for (int j = 0; j < 8; ++j) a[8 + j] = row[kbase + 16 + hi * 8 + j];
    return a;
}

// B fragment (32x16 bf16) from chunk-transposed storage bt[n][kk] (kk=0..31).
// Lane l: n = l&15, hi = l>>4.  elems j<8: K = hi*8 + j; elems j>=8: K = 16 + hi*8 + (j-8).
__device__ __forceinline__ v16bf b_frag(const bf16_t* bt_n_row, int hi) {
    v16bf b;
#pragma unroll
    for (int j = 0; j < 8; ++j) b[j] = bt_n_row[hi * 8 + j];
#pragma unroll
    for (int j = 0; j < 8; ++j) b[8 + j] = bt_n_row[16 + hi * 8 + j];
    return b;
}

__device__ __forceinline__ float leaky(float v) { return v > 0.0f ? v : LRELU * v; }

// ---------------------------------------------------------------------------
// Utility kernels
// ---------------------------------------------------------------------------
__global__ void zero_kernel(float* __restrict__ p, size_t n) {
    size_t i = (size_t)blockIdx.x * blockDim.x + threadIdx.x;
    if (i < n) p[i] = 0.0f;
}

__global__ void counts_kernel(const int* __restrict__ dst, const int* __restrict__ batch,
                              float* __restrict__ deg, float* __restrict__ cnt) {
    int i = blockIdx.x * blockDim.x + threadIdx.x;
    if (i < Ee) unsafeAtomicAdd(&deg[dst[i]], 1.0f);
    if (i < Nn) unsafeAtomicAdd(&cnt[batch[i]], 1.0f);
}

// src f32 [K x Nc] row-major -> dst bf16 [K/32][Nc][32]  (chunk-transposed)
__global__ void convert_chunked_kernel(const float* __restrict__ src, bf16_t* __restrict__ dst,
                                       int K, int Nc) {
    int i = blockIdx.x * blockDim.x + threadIdx.x;
    if (i >= K * Nc) return;
    int kk = i & 31;
    int rest = i >> 5;
    int n = rest % Nc;
    int c = rest / Nc;
    dst[i] = (bf16_t)src[(c * 32 + kk) * Nc + n];
}

// W2 [128, 4096] (+ b2 [4096] appended as 64 extra K rows) -> bf16 [258][64][32]
__global__ void convert_w2t_kernel(const float* __restrict__ W2, const float* __restrict__ b2,
                                   bf16_t* __restrict__ w2t) {
    int i = blockIdx.x * blockDim.x + threadIdx.x;
    if (i >= 258 * 64 * 32) return;
    int kk = i & 31;
    int rest = i >> 5;
    int n = rest & 63;
    int c = rest >> 6;
    int r = c * 32 + kk;
    float v;
    if (r < 8192) {
        int k = r >> 6, d = r & 63;
        v = W2[k * 4096 + d * 64 + n];
    } else {
        v = b2[(r - 8192) * 64 + n];
    }
    w2t[i] = (bf16_t)v;
}

// ---------------------------------------------------------------------------
// Per-layer: LayerNorm + leaky -> h (bf16).  One wave per node, 2 elems/lane.
// ---------------------------------------------------------------------------
__global__ __launch_bounds__(256) void prep_layer_kernel(const float* __restrict__ x,
                                                         const float* __restrict__ ln_scale,
                                                         const float* __restrict__ ln_bias,
                                                         bf16_t* __restrict__ h_bf) {
    int wave = (blockIdx.x * blockDim.x + threadIdx.x) >> 5;
    int lane = threadIdx.x & 31;
    if (wave >= Nn) return;
    const float* row = x + (size_t)wave * 64;
    float a0 = row[lane], a1 = row[lane + 32];
    float s = a0 + a1;
#pragma unroll
    for (int m = 16; m >= 1; m >>= 1) s += __shfl_xor(s, m, 32);
    float mu = s * (1.0f / 64.0f);
    float d0 = a0 - mu, d1 = a1 - mu;
    float v = d0 * d0 + d1 * d1;
#pragma unroll
    for (int m = 16; m >= 1; m >>= 1) v += __shfl_xor(v, m, 32);
    float rstd = rsqrtf(v * (1.0f / 64.0f) + LN_EPS);
    float h0 = leaky(d0 * rstd * ln_scale[lane] + ln_bias[lane]);
    float h1 = leaky(d1 * rstd * ln_scale[lane + 32] + ln_bias[lane + 32]);
    h_bf[(size_t)wave * 64 + lane]      = (bf16_t)h0;
    h_bf[(size_t)wave * 64 + lane + 32] = (bf16_t)h1;
}

// ---------------------------------------------------------------------------
// Edge MLP stage 1: u = leaky(edge_attr @ W1 + b1)  [E,64]@[64,128] -> bf16
// ---------------------------------------------------------------------------
__global__ __launch_bounds__(256) void edge_mlp1_kernel(const float* __restrict__ edge_attr,
                                                        const bf16_t* __restrict__ w1t,
                                                        const float* __restrict__ b1,
                                                        bf16_t* __restrict__ u_bf) {
    __shared__ bf16_t ea[128][72];
    __shared__ bf16_t w1s[2 * 128 * 32];
    int t = threadIdx.x;
    int eb = blockIdx.x * 128;
    {
        int e = t >> 1, half = t & 1;
        const float* sp = edge_attr + (size_t)(eb + e) * 64 + half * 32;
#pragma unroll
        for (int i = 0; i < 32; ++i) ea[e][half * 32 + i] = (bf16_t)sp[i];
    }
    {
        const uint4* s = (const uint4*)(w1t) + t * 4;  // 64B/thread, 16KB total
        uint4* d = (uint4*)(w1s) + t * 4;
#pragma unroll
        for (int i = 0; i < 4; ++i) d[i] = s[i];
    }
    __syncthreads();

    int wave = t >> 5, lane = t & 31, mi = lane & 15, hi = lane >> 4;
    int mbase = wave * 16;
    v8f acc[8] = {};
#pragma unroll
    for (int c = 0; c < 2; ++c) {
        v16bf a = a_frag_row(&ea[mbase + mi][0], c * 32, hi);
#pragma unroll
        for (int nt = 0; nt < 8; ++nt) {
            v16bf b = b_frag(&w1s[(c * 128 + nt * 16 + mi) * 32], hi);
            acc[nt] = wmma_bf16(a, b, acc[nt]);
        }
    }
#pragma unroll
    for (int nt = 0; nt < 8; ++nt) {
        int col = nt * 16 + mi;
        float bias = b1[col];
#pragma unroll
        for (int v = 0; v < 8; ++v) {
            int e = eb + mbase + hi * 8 + v;
            u_bf[(size_t)e * 128 + col] = (bf16_t)leaky(acc[nt][v] + bias);
        }
    }
}

// ---------------------------------------------------------------------------
// Fused edge message GEMM: msg[e] = (u[e] (x) h[src[e]]) @ W2ext -> atomic num.
// K = 258 chunks of 32 = 43 stages x 6 chunks; B double-buffered in LDS via
// global_load_async_to_lds_b128, one stage ahead; 24 WMMAs per barrier pair.
// ---------------------------------------------------------------------------
#define MSG_SB 6                    // K-chunks per stage
#define MSG_STAGES 43               // 43 * 6 = 258
#define MSG_STAGE_BYTES (MSG_SB * 4096)

__global__ __launch_bounds__(256) void edge_msg_kernel(const bf16_t* __restrict__ h_bf,
                                                       const bf16_t* __restrict__ u_bf,
                                                       const bf16_t* __restrict__ w2t,
                                                       const int* __restrict__ src,
                                                       const int* __restrict__ dst,
                                                       float* __restrict__ num) {
    __shared__ bf16_t hs[128][72];
    __shared__ bf16_t ul[128][136];
    __shared__ __align__(16) bf16_t bs[2][MSG_SB * 2048];
    int t = threadIdx.x;
    int eb = blockIdx.x * 128;
    {   // gather h[src] rows (32 bf16 / thread, vectorized)
        int e = t >> 1, half = t & 1;
        int s = src[eb + e];
        const uint4* sp = (const uint4*)(h_bf + (size_t)s * 64 + half * 32);
        uint4* dp = (uint4*)(&hs[e][half * 32]);
#pragma unroll
        for (int i = 0; i < 4; ++i) dp[i] = sp[i];
    }
    {   // load u rows (64 bf16 / thread)
        int e = t >> 1, half = t & 1;
        const uint4* sp = (const uint4*)(u_bf + (size_t)(eb + e) * 128 + half * 64);
        uint4* dp = (uint4*)(&ul[e][half * 64]);
#pragma unroll
        for (int i = 0; i < 8; ++i) dp[i] = sp[i];
    }

    // Per-thread async staging bases: thread t owns 16B at offset t*16 of each 4KB slab.
    unsigned lds_b0 = lds_off_of(&bs[0][0]) + t * 16;
    unsigned lds_b1 = lds_off_of(&bs[1][0]) + t * 16;
    const char* gw2 = (const char*)w2t + t * 16;

    // Prologue: async-stage stage 0 into buffer 0.
#pragma unroll
    for (int i = 0; i < MSG_SB; ++i)
        async_copy_b128(lds_b0 + i * 4096, gw2 + i * 4096);

    int wave = t >> 5, lane = t & 31, mi = lane & 15, hi = lane >> 4;
    int mbase = wave * 16;
    v8f acc[4] = {};

    for (int s = 0; s < MSG_STAGES; ++s) {
        if (s + 1 < MSG_STAGES) {
            // Prefetch next stage into the other buffer (read-safe: last read of
            // that buffer completed before the trailing barrier of iter s-1).
            unsigned lb = ((s + 1) & 1) ? lds_b1 : lds_b0;
            const char* g = gw2 + (size_t)(s + 1) * MSG_STAGE_BYTES;
#pragma unroll
            for (int i = 0; i < MSG_SB; ++i)
                async_copy_b128(lb + i * 4096, g + i * 4096);
            asm volatile("s_wait_asynccnt 6" ::: "memory");   // stage s complete
        } else {
            asm volatile("s_wait_asynccnt 0" ::: "memory");
        }
        __syncthreads();  // all waves' stage-s data visible (also covers hs/ul on s==0)

        const bf16_t* bstage = &bs[s & 1][0];
        const bf16_t* hrow = &hs[mbase + mi][0];
#pragma unroll
        for (int q = 0; q < MSG_SB; ++q) {
            int c = s * MSG_SB + q;
            v16bf a;
            if (c < 256) {
                int k = c >> 1;
                int dd0 = (c & 1) * 32;
                float us = (float)ul[mbase + mi][k];
#pragma unroll
                for (int j = 0; j < 8; ++j)
                    a[j] = (bf16_t)(us * (float)hrow[dd0 + hi * 8 + j]);
#pragma unroll
                for (int j = 0; j < 8; ++j)
                    a[8 + j] = (bf16_t)(us * (float)hrow[dd0 + 16 + hi * 8 + j]);
            } else {
                a = a_frag_row(hrow, (c - 256) * 32, hi);  // b2 rows: A = h[src]
            }
#pragma unroll
            for (int nt = 0; nt < 4; ++nt) {
                v16bf b = b_frag(&bstage[q * 2048 + (nt * 16 + mi) * 32], hi);
                acc[nt] = wmma_bf16(a, b, acc[nt]);
            }
        }
        __syncthreads();  // all reads of buffer (s&1) done before it is re-staged
    }

    int dste[8];
#pragma unroll
    for (int v = 0; v < 8; ++v) dste[v] = dst[eb + mbase + hi * 8 + v];
#pragma unroll
    for (int nt = 0; nt < 4; ++nt) {
        int col = nt * 16 + mi;
#pragma unroll
        for (int v = 0; v < 8; ++v)
            unsafeAtomicAdd(&num[(size_t)dste[v] * 64 + col], acc[nt][v]);
    }
}

// ---------------------------------------------------------------------------
// Layer epilogue: x += num/max(deg,1) + h@root + conv_b; re-zero num.
// ---------------------------------------------------------------------------
__global__ __launch_bounds__(256) void finish_layer_kernel(float* __restrict__ x_cur,
                                                           const bf16_t* __restrict__ h_bf,
                                                           const bf16_t* __restrict__ roott,
                                                           const float* __restrict__ conv_b,
                                                           float* __restrict__ num,
                                                           const float* __restrict__ deg) {
    __shared__ bf16_t ht[128][72];
    __shared__ bf16_t rs[2 * 64 * 32];
    int t = threadIdx.x;
    int nb = blockIdx.x * 128;
    {
        int n = t >> 1, half = t & 1;
        const uint4* sp = (const uint4*)(h_bf + (size_t)(nb + n) * 64 + half * 32);
        uint4* dp = (uint4*)(&ht[n][half * 32]);
#pragma unroll
        for (int i = 0; i < 4; ++i) dp[i] = sp[i];
    }
    {
        const uint4* s = (const uint4*)(roott) + t * 2;
        uint4* d = (uint4*)(rs) + t * 2;
        d[0] = s[0]; d[1] = s[1];
    }
    __syncthreads();

    int wave = t >> 5, lane = t & 31, mi = lane & 15, hi = lane >> 4;
    int mbase = wave * 16;
    v8f acc[4] = {};
#pragma unroll
    for (int c = 0; c < 2; ++c) {
        v16bf a = a_frag_row(&ht[mbase + mi][0], c * 32, hi);
#pragma unroll
        for (int nt = 0; nt < 4; ++nt) {
            v16bf b = b_frag(&rs[(c * 64 + nt * 16 + mi) * 32], hi);
            acc[nt] = wmma_bf16(a, b, acc[nt]);
        }
    }
    float inv[8];
#pragma unroll
    for (int v = 0; v < 8; ++v)
        inv[v] = 1.0f / fmaxf(deg[nb + mbase + hi * 8 + v], 1.0f);
#pragma unroll
    for (int nt = 0; nt < 4; ++nt) {
        int col = nt * 16 + mi;
        float cb = conv_b[col];
#pragma unroll
        for (int v = 0; v < 8; ++v) {
            size_t idx = (size_t)(nb + mbase + hi * 8 + v) * 64 + col;
            x_cur[idx] = x_cur[idx] + num[idx] * inv[v] + acc[nt][v] + cb;
            num[idx] = 0.0f;  // ready for next layer
        }
    }
}

// ---------------------------------------------------------------------------
// Head: dense = x@dense_W + dense_b, atomically pooled per graph.
// ---------------------------------------------------------------------------
__global__ __launch_bounds__(256) void head_dense_kernel(const float* __restrict__ x_cur,
                                                         const bf16_t* __restrict__ denset,
                                                         const float* __restrict__ dense_b,
                                                         const int* __restrict__ batch,
                                                         float* __restrict__ pooled) {
    __shared__ bf16_t xt[128][72];
    __shared__ bf16_t ds[2 * 64 * 32];
    int t = threadIdx.x;
    int nb = blockIdx.x * 128;
    {
        int n = t >> 1, half = t & 1;
        const float* sp = x_cur + (size_t)(nb + n) * 64 + half * 32;
#pragma unroll
        for (int i = 0; i < 32; ++i) xt[n][half * 32 + i] = (bf16_t)sp[i];
    }
    {
        const uint4* s = (const uint4*)(denset) + t * 2;
        uint4* d = (uint4*)(ds) + t * 2;
        d[0] = s[0]; d[1] = s[1];
    }
    __syncthreads();

    int wave = t >> 5, lane = t & 31, mi = lane & 15, hi = lane >> 4;
    int mbase = wave * 16;
    v8f acc[4] = {};
#pragma unroll
    for (int c = 0; c < 2; ++c) {
        v16bf a = a_frag_row(&xt[mbase + mi][0], c * 32, hi);
#pragma unroll
        for (int nt = 0; nt < 4; ++nt) {
            v16bf b = b_frag(&ds[(c * 64 + nt * 16 + mi) * 32], hi);
            acc[nt] = wmma_bf16(a, b, acc[nt]);
        }
    }
    int bg[8];
#pragma unroll
    for (int v = 0; v < 8; ++v) bg[v] = batch[nb + mbase + hi * 8 + v];
#pragma unroll
    for (int nt = 0; nt < 4; ++nt) {
        int col = nt * 16 + mi;
        float db = dense_b[col];
#pragma unroll
        for (int v = 0; v < 8; ++v)
            unsafeAtomicAdd(&pooled[(size_t)bg[v] * 64 + col], acc[nt][v] + db);
    }
}

// ---------------------------------------------------------------------------
// Final: out = leaky(pooled/cnt) @ out_W + out_b   [512,64]@[64,128]
// ---------------------------------------------------------------------------
__global__ __launch_bounds__(256) void final_out_kernel(const float* __restrict__ pooled,
                                                        const float* __restrict__ cnt,
                                                        const bf16_t* __restrict__ outt,
                                                        const float* __restrict__ out_b,
                                                        float* __restrict__ out) {
    __shared__ bf16_t pt[128][72];
    __shared__ bf16_t os[2 * 128 * 32];
    int t = threadIdx.x;
    int gb = blockIdx.x * 128;
    {
        int g = t >> 1, half = t & 1;
        float ic = 1.0f / fmaxf(cnt[gb + g], 1.0f);
        const float* sp = pooled + (size_t)(gb + g) * 64 + half * 32;
#pragma unroll
        for (int i = 0; i < 32; ++i) pt[g][half * 32 + i] = (bf16_t)leaky(sp[i] * ic);
    }
    {
        const uint4* s = (const uint4*)(outt) + t * 4;
        uint4* d = (uint4*)(os) + t * 4;
#pragma unroll
        for (int i = 0; i < 4; ++i) d[i] = s[i];
    }
    __syncthreads();

    int wave = t >> 5, lane = t & 31, mi = lane & 15, hi = lane >> 4;
    int mbase = wave * 16;
    v8f acc[8] = {};
#pragma unroll
    for (int c = 0; c < 2; ++c) {
        v16bf a = a_frag_row(&pt[mbase + mi][0], c * 32, hi);
#pragma unroll
        for (int nt = 0; nt < 8; ++nt) {
            v16bf b = b_frag(&os[(c * 128 + nt * 16 + mi) * 32], hi);
            acc[nt] = wmma_bf16(a, b, acc[nt]);
        }
    }
#pragma unroll
    for (int nt = 0; nt < 8; ++nt) {
        int col = nt * 16 + mi;
        float ob = out_b[col];
#pragma unroll
        for (int v = 0; v < 8; ++v) {
            int g = gb + mbase + hi * 8 + v;
            out[(size_t)g * 128 + col] = acc[nt][v] + ob;
        }
    }
}

// ---------------------------------------------------------------------------
extern "C" void kernel_launch(void* const* d_in, const int* in_sizes, int n_in,
                              void* d_out, int out_size, void* d_ws, size_t ws_size,
                              hipStream_t stream) {
    (void)in_sizes; (void)n_in; (void)out_size; (void)ws_size;
    const float* x         = (const float*)d_in[0];
    const int*   src       = (const int*)d_in[1];
    const int*   dst       = (const int*)d_in[1] + Ee;
    const float* edge_attr = (const float*)d_in[2];
    const int*   batch     = (const int*)d_in[3];
    const float* ln_scale  = (const float*)d_in[4];
    const float* ln_bias   = (const float*)d_in[5];
    const float* W1        = (const float*)d_in[6];
    const float* b1        = (const float*)d_in[7];
    const float* W2        = (const float*)d_in[8];
    const float* b2        = (const float*)d_in[9];
    const float* root      = (const float*)d_in[10];
    const float* conv_b    = (const float*)d_in[11];
    const float* dense_W   = (const float*)d_in[12];
    const float* dense_b   = (const float*)d_in[13];
    const float* out_W     = (const float*)d_in[14];
    const float* out_b     = (const float*)d_in[15];

    size_t off = 0;
    auto take = [&](size_t bytes) -> void* {
        void* p = (char*)d_ws + off;
        off = (off + bytes + 255) & ~(size_t)255;
        return p;
    };
    // Contiguous zeroed region first (all sizes are multiples of 256B):
    float*  num_ws = (float*)take((size_t)Nn * 64 * 4);
    float*  deg_ws = (float*)take((size_t)Nn * 4);
    float*  pooled = (float*)take((size_t)Gg * 64 * 4);
    float*  cnt    = (float*)take((size_t)Gg * 4);
    float*  x_cur  = (float*)take((size_t)Nn * 64 * 4);
    bf16_t* h_bf   = (bf16_t*)take((size_t)Nn * 64 * 2);
    bf16_t* u_bf   = (bf16_t*)take((size_t)Ee * 128 * 2);
    bf16_t* w1t    = (bf16_t*)take((size_t)Ll * 2 * 128 * 32 * 2);
    bf16_t* w2t    = (bf16_t*)take((size_t)Ll * 258 * 64 * 32 * 2);
    bf16_t* roott  = (bf16_t*)take((size_t)Ll * 2 * 64 * 32 * 2);
    bf16_t* denset = (bf16_t*)take((size_t)2 * 64 * 32 * 2);
    bf16_t* outt   = (bf16_t*)take((size_t)2 * 128 * 32 * 2);

    const size_t zero_n = (size_t)Nn * 64 + Nn + (size_t)Gg * 64 + Gg;
    zero_kernel<<<(unsigned)((zero_n + 255) / 256), 256, 0, stream>>>(num_ws, zero_n);
    hipMemcpyAsync(x_cur, x, (size_t)Nn * 64 * sizeof(float), hipMemcpyDeviceToDevice, stream);
    counts_kernel<<<Ee / 256, 256, 0, stream>>>(dst, batch, deg_ws, cnt);

    for (int l = 0; l < Ll; ++l) {
        convert_chunked_kernel<<<(64 * 128) / 256, 256, 0, stream>>>(
            W1 + (size_t)l * Dd * Hh, w1t + (size_t)l * 8192, 64, 128);
        convert_w2t_kernel<<<(258 * 64 * 32 + 255) / 256, 256, 0, stream>>>(
            W2 + (size_t)l * Hh * Dd * Dd, b2 + (size_t)l * Dd * Dd, w2t + (size_t)l * 528384);
        convert_chunked_kernel<<<(64 * 64) / 256, 256, 0, stream>>>(
            root + (size_t)l * Dd * Dd, roott + (size_t)l * 4096, 64, 64);
    }
    convert_chunked_kernel<<<(64 * 64) / 256, 256, 0, stream>>>(dense_W, denset, 64, 64);
    convert_chunked_kernel<<<(64 * 128) / 256, 256, 0, stream>>>(out_W, outt, 64, 128);

    for (int l = 0; l < Ll; ++l) {
        prep_layer_kernel<<<Nn / 8, 256, 0, stream>>>(
            x_cur, ln_scale + (size_t)l * Dd, ln_bias + (size_t)l * Dd, h_bf);
        edge_mlp1_kernel<<<Ee / 128, 256, 0, stream>>>(
            edge_attr, w1t + (size_t)l * 8192, b1 + (size_t)l * Hh, u_bf);
        edge_msg_kernel<<<Ee / 128, 256, 0, stream>>>(
            h_bf, u_bf, w2t + (size_t)l * 528384, src, dst, num_ws);
        finish_layer_kernel<<<Nn / 128, 256, 0, stream>>>(
            x_cur, h_bf, roott + (size_t)l * 4096, conv_b + (size_t)l * Dd, num_ws, deg_ws);
    }

    head_dense_kernel<<<Nn / 128, 256, 0, stream>>>(x_cur, denset, dense_b, batch, pooled);
    final_out_kernel<<<Gg / 128, 256, 0, stream>>>(pooled, cnt, outt, out_b, (float*)d_out);
}
